// Attention_69836168233316
// MI455X (gfx1250) — matrix operand order repaired and checked by reading, not compiled, not run on hardware
//
#include <hip/hip_runtime.h>
#include <hip/hip_bf16.h>
#include <cstdint>

// ---------------------------------------------------------------------------
// Problem constants
// ---------------------------------------------------------------------------
#define BB    8
#define NN    1024
#define DIMM  512
#define HH    8
#define DHH   64
#define INNR  (HH * DHH)          // 512
#define MROWS (BB * NN)           // 8192
#define SCALE 0.125f              // DH^-0.5

typedef __attribute__((ext_vector_type(16))) __bf16 v16bf;
typedef __attribute__((ext_vector_type(8)))  float  v8f;
typedef __attribute__((ext_vector_type(4)))  unsigned int v4u;
typedef __attribute__((ext_vector_type(8)))  int v8i;
typedef __attribute__((ext_vector_type(4)))  int v4i;

// GCC-style vector to match the async-LDS builtin's parameter type exactly:
// "__attribute__((__vector_size__(4 * sizeof(int)))) int __device__ *" (non-const!)
typedef int vsi4 __attribute__((vector_size(16)));
typedef __attribute__((address_space(1))) vsi4 as1_vsi4;   // global (device) memory
typedef __attribute__((address_space(3))) vsi4 as3_vsi4;   // LDS

// CDNA5 feature guards (device pass only; host pass sees the fallbacks)
#define HAS_ASYNC_LDS __has_builtin(__builtin_amdgcn_global_load_async_to_lds_b128)
#define HAS_TDM       __has_builtin(__builtin_amdgcn_tensor_load_to_lds)

__device__ __forceinline__ as1_vsi4* to_global_v4(const void* p) {
  return (as1_vsi4*)(uintptr_t)p;            // generic global addr == AS1 addr (const dropped)
}
__device__ __forceinline__ as3_vsi4* to_lds_v4(const void* p) {
  return (as3_vsi4*)(uint32_t)(uintptr_t)p;  // low 32 bits of generic LDS ptr = LDS offset
}

__device__ __forceinline__ void wait_async0() {
#if __has_builtin(__builtin_amdgcn_s_wait_asynccnt)
  __builtin_amdgcn_s_wait_asynccnt(0);
#else
  asm volatile("s_wait_asynccnt 0" ::: "memory");
#endif
}

// ---------------------------------------------------------------------------
// WMMA helpers: CDNA5 v_wmma_f32_16x16x32_bf16 fragment builders.
// A (16x32, 16-bit): lane L holds row M=L%16; lanes 0-15: K {0..7,16..23},
// lanes 16-31: K {8..15,24..31}; VGPR v holds a K pair.
// B (32x16, 16-bit): lane L holds col N=L%16; same K order as A.
// C/D (16x16 f32): lane L col N=L%16; element r is row M = r + 8*(L/16).
// ---------------------------------------------------------------------------
__device__ __forceinline__ int kmapA(int v, int hi) {
  return ((v & 4) << 2) + hi * 8 + ((v & 3) << 1);   // {0,16} + 8*hi + 2*(v%4)
}

__device__ __forceinline__ v16bf make_a_f32(const float* __restrict__ row, int hi, int kofs) {
  v16bf a;
#pragma unroll
  for (int v = 0; v < 8; ++v) {
    int k = kofs + kmapA(v, hi);
    a[2 * v]     = (__bf16)row[k];
    a[2 * v + 1] = (__bf16)row[k + 1];
  }
  return a;
}

__device__ __forceinline__ v16bf make_a_bf16(const __bf16* __restrict__ row, int hi, int kofs) {
  v16bf a;
#pragma unroll
  for (int v = 0; v < 8; ++v) {
    int k = kofs + kmapA(v, hi);
    a[2 * v]     = row[k];
    a[2 * v + 1] = row[k + 1];
  }
  return a;
}

// base points at element (k=0, this lane's column); element (k,n) = base[k*stride]
__device__ __forceinline__ v16bf make_b_bf16(const __bf16* __restrict__ base, int khalf,
                                             int kofs, int stride) {
  v16bf b;
#pragma unroll
  for (int v = 0; v < 8; ++v) {
    int k = kofs + kmapA(v, khalf);
    b[2 * v]     = base[(size_t)k * stride];
    b[2 * v + 1] = base[(size_t)(k + 1) * stride];
  }
  return b;
}

__device__ __forceinline__ v8f wmma_bf16(v16bf a, v16bf b, v8f c) {
  return __builtin_amdgcn_wmma_f32_16x16x32_bf16(false, a, false, b, (short)0, c, false, false);
}

// ---------------------------------------------------------------------------
// Kernel 0: convert weights f32 -> bf16 into workspace
// ---------------------------------------------------------------------------
#define WQ_ELEMS (3 * INNR * DIMM)   // 786432
#define WM_ELEMS (DHH * DHH)         // 4096
#define WO_ELEMS (DIMM * INNR)       // 262144

__global__ void cvt_weights_kernel(const float* __restrict__ wq, const float* __restrict__ wm,
                                   const float* __restrict__ wo, __bf16* __restrict__ wqb,
                                   __bf16* __restrict__ wmb, __bf16* __restrict__ wob) {
  const int total = WQ_ELEMS + WM_ELEMS + WO_ELEMS;
  for (int i = blockIdx.x * blockDim.x + threadIdx.x; i < total; i += gridDim.x * blockDim.x) {
    if (i < WQ_ELEMS) {
      wqb[i] = (__bf16)wq[i];
    } else if (i < WQ_ELEMS + WM_ELEMS) {
      wmb[i - WQ_ELEMS] = (__bf16)wm[i - WQ_ELEMS];
    } else {
      wob[i - WQ_ELEMS - WM_ELEMS] = (__bf16)wo[i - WQ_ELEMS - WM_ELEMS];
    }
  }
}

// ---------------------------------------------------------------------------
// Kernel 1: split lidar into heads, f32 -> bf16, [B,N,H*DH] -> [B,H,N,DH]
// ---------------------------------------------------------------------------
__global__ void split_lidar_kernel(const float* __restrict__ lidar, __bf16* __restrict__ lidh) {
  const int total = BB * NN * INNR;
  for (int i = blockIdx.x * blockDim.x + threadIdx.x; i < total; i += gridDim.x * blockDim.x) {
    int b = i >> 19;               // / (NN*INNR)
    int rem = i & ((1 << 19) - 1);
    int n = rem >> 9;              // / INNR
    int inner = rem & 511;
    int h = inner >> 6, d = inner & 63;
    lidh[(size_t)((b * HH + h) * NN + n) * DHH + d] = (__bf16)lidar[i];
  }
}

// ---------------------------------------------------------------------------
// Kernel 2: QKV projection.  qkv[m,c] = sum_k x[m,k] * w_qkv[c,k]
// ---------------------------------------------------------------------------
__global__ void qkv_gemm_kernel(const float* __restrict__ x, const __bf16* __restrict__ wqb,
                                __bf16* __restrict__ qh, __bf16* __restrict__ kh,
                                __bf16* __restrict__ vh) {
  const int wid  = blockIdx.x * 8 + (threadIdx.x >> 5);
  const int lane = threadIdx.x & 31;
  const int hi   = lane >> 4;
  const int l16  = lane & 15;
  const int tm   = wid / 96;        // 512 row tiles
  const int tn   = wid % 96;        // 96 col tiles of 1536

  const float*  arow = x   + (size_t)(tm * 16 + l16) * DIMM;
  const __bf16* brow = wqb + (size_t)(tn * 16 + l16) * DIMM;

  v8f acc = {};
#pragma unroll 4
  for (int kk = 0; kk < DIMM; kk += 32) {
    if (kk + 64 < DIMM) __builtin_prefetch(arow + kk + 64, 0, 1);
    v16bf a = make_a_f32(arow, hi, kk);
    v16bf b = make_b_bf16(brow, hi, kk, 1);
    acc = wmma_bf16(a, b, acc);
  }

  const int c   = tn * 16 + l16;
  const int sec = c >> 9;           // 0=q 1=k 2=v
  const int cin = c & 511;
  const int h   = cin >> 6, d = cin & 63;
  __bf16* dst = (sec == 0) ? qh : (sec == 1) ? kh : vh;
#pragma unroll
  for (int r = 0; r < 8; ++r) {
    int m  = tm * 16 + r + 8 * hi;
    int bb = m >> 10, nn = m & 1023;
    dst[(size_t)((bb * HH + h) * NN + nn) * DHH + d] = (__bf16)acc[r];
  }
}

// ---------------------------------------------------------------------------
// Kernel 3: fused attention. One block = one (b, h, 32-row strip).
// Dynamic LDS (fixed 272KB layout, independent of feature guards):
//   S[32][1024] f32 | O[32][64] f32 | V[1024][64] bf16 | lidA[32][64] | qA[32][64]
// ---------------------------------------------------------------------------
#define S_FLOATS (32 * NN)
#define O_FLOATS (32 * DHH)
#define V_ELEMS  (NN * DHH)
#define AR_ELEMS (32 * DHH)
#define ATTN_SMEM_BYTES ((S_FLOATS + O_FLOATS) * 4 + (V_ELEMS + 2 * AR_ELEMS) * 2)  // 278528

__device__ __forceinline__ void softmax_rows_32x1024(float* __restrict__ S, float mulv,
                                                     float addv) {
  const int tid = threadIdx.x;
  const int row = tid >> 3, sub = tid & 7;
  float* r = S + row * NN;
  float mx = -3.402823466e38f;
  for (int c = sub; c < NN; c += 8) mx = fmaxf(mx, r[c]);
  mx = fmaxf(mx, __shfl_xor(mx, 1, 8));
  mx = fmaxf(mx, __shfl_xor(mx, 2, 8));
  mx = fmaxf(mx, __shfl_xor(mx, 4, 8));
  float s = 0.f;
  for (int c = sub; c < NN; c += 8) s += __expf(r[c] - mx);
  s += __shfl_xor(s, 1, 8);
  s += __shfl_xor(s, 2, 8);
  s += __shfl_xor(s, 4, 8);
  const float inv = 1.0f / s;
  for (int c = sub; c < NN; c += 8) r[c] = mulv * (__expf(r[c] - mx) * inv) + addv;
}

__global__ void attn_kernel(const __bf16* __restrict__ qh, const __bf16* __restrict__ kh,
                            const __bf16* __restrict__ vh, const __bf16* __restrict__ lidh,
                            const __bf16* __restrict__ wmb, const float* __restrict__ bmerge,
                            const float* __restrict__ convw, const float* __restrict__ convb,
                            __bf16* __restrict__ merged) {
  extern __shared__ float smem[];
  float*  S    = smem;                           // 32 x 1024 f32
  float*  O    = smem + S_FLOATS;                // 32 x 64 f32
  __bf16* Vl   = (__bf16*)(O + O_FLOATS);        // 1024 x 64 bf16 (TDM target)
  __bf16* lidA = Vl + V_ELEMS;                   // 32 x 64 bf16
  __bf16* qA   = lidA + AR_ELEMS;                // 32 x 64 bf16

  const int bh = blockIdx.x >> 5;   // 0..63 = b*8+h
  const int ib = blockIdx.x & 31;
  const int i0 = ib * 32;

  const size_t bhofs = (size_t)bh * NN * DHH;
  const __bf16* qbh = qh   + bhofs;
  const __bf16* kbh = kh   + bhofs;
  const __bf16* vbh = vh   + bhofs;
  const __bf16* lbh = lidh + bhofs;

  const int wave = threadIdx.x >> 5;
  const int lane = threadIdx.x & 31;
  const int hi   = lane >> 4;
  const int l16  = lane & 15;
  const int rh   = wave & 1;
  const int q4   = wave >> 1;       // j-quarter (A/B) and d-quarter (C/D)
  const int rbase = rh * 16;

  const float w0 = convw[0], w1 = convw[1], cb = convb[0];

  // ---- TDM: DMA the full 1024x64 bf16 V panel into LDS (wave 0 issues) -----
#if HAS_TDM
  if (wave == 0) {
    uint64_t ga = (uint64_t)(uintptr_t)vbh;       // global byte address of tile
    uint32_t la = (uint32_t)(uintptr_t)Vl;        // LDS byte offset
    v4u g0;
    g0.x = 1u;                                            // count=1, user mode
    g0.y = la;                                            // lds_addr
    g0.z = (uint32_t)ga;                                  // global_addr[31:0]
    g0.w = ((uint32_t)(ga >> 32) & 0x01FFFFFFu) | 0x80000000u;  // addr[56:32] | type=2
    v8i g1;
    g1[0] = 1 << 16;          // workgroup_mask=0, data_size=1 (2 bytes)
    g1[1] = 64 << 16;         // tensor_dim0 = 64 (low16 at bits 63:48)
    g1[2] = 1024 << 16;       // tensor_dim0 hi=0 | tensor_dim1 low16 = 1024
    g1[3] = 64 << 16;         // tensor_dim1 hi=0 | tile_dim0 = 64
    g1[4] = 1024;             // tile_dim1 = 1024 | tile_dim2 = 0
    g1[5] = 64;               // tensor_dim0_stride[31:0] = 64
    g1[6] = 0;                // stride0 hi | stride1 low16 (65536 -> 0)
    g1[7] = 1;                // stride1[47:16] = 65536>>16 = 1
    v4i z = {0, 0, 0, 0};
#if __clang_major__ >= 23
    v8i z8 = {0, 0, 0, 0, 0, 0, 0, 0};
    __builtin_amdgcn_tensor_load_to_lds(g0, g1, z, z, z8, 0);
#else
    __builtin_amdgcn_tensor_load_to_lds(g0, g1, z, z, 0);
#endif
  }
#endif

  // ---- Async-stage the q / lidar A-row strips (8KB each) into LDS ----------
  {
    const __bf16* gsrcL = lbh + (size_t)i0 * DHH;
    const __bf16* gsrcQ = qbh + (size_t)i0 * DHH;
    const int t = threadIdx.x;
#pragma unroll
    for (int c = 0; c < 2; ++c) {
      int off = (t + c * 256) * 8;   // 8 bf16 = 16 bytes per transfer
#if HAS_ASYNC_LDS
      __builtin_amdgcn_global_load_async_to_lds_b128(to_global_v4(gsrcL + off),
                                                     to_lds_v4(lidA + off), 0, 0);
      __builtin_amdgcn_global_load_async_to_lds_b128(to_global_v4(gsrcQ + off),
                                                     to_lds_v4(qA + off), 0, 0);
#else
      *(float4*)(lidA + off) = *(const float4*)(gsrcL + off);
      *(float4*)(qA + off)   = *(const float4*)(gsrcQ + off);
#endif
    }
#if HAS_ASYNC_LDS
    wait_async0();
#endif
  }
  __syncthreads();

  // ---- Phase A: S = SCALE * (lid . lid^T) strip ----------------------------
  {
    const __bf16* arow = lidA + (rbase + l16) * DHH;     // LDS
    v16bf la0 = make_a_bf16(arow, hi, 0);
    v16bf la1 = make_a_bf16(arow, hi, 32);
#pragma unroll 2
    for (int jt = 0; jt < 16; ++jt) {
      int jbase = q4 * 256 + jt * 16;
      const __bf16* bcol = lbh + (size_t)(jbase + l16) * DHH;
      v16bf b0 = make_b_bf16(bcol, hi, 0, 1);
      v16bf b1 = make_b_bf16(bcol, hi, 32, 1);
      v8f acc = {};
      acc = wmma_bf16(la0, b0, acc);
      acc = wmma_bf16(la1, b1, acc);
#pragma unroll
      for (int r = 0; r < 8; ++r)
        S[(rbase + r + 8 * hi) * NN + jbase + l16] = acc[r] * SCALE;
    }
  }
  __syncthreads();

  // ---- lidar softmax folded into blend: S = w1*softmax(S) + conv_b ---------
  softmax_rows_32x1024(S, w1, cb);
  __syncthreads();

  // ---- Phase B: S += w0 * SCALE * (q . k^T) --------------------------------
  {
    const __bf16* arow = qA + (rbase + l16) * DHH;       // LDS
    v16bf qa0 = make_a_bf16(arow, hi, 0);
    v16bf qa1 = make_a_bf16(arow, hi, 32);
    const float blend = w0 * SCALE;
#pragma unroll 2
    for (int jt = 0; jt < 16; ++jt) {
      int jbase = q4 * 256 + jt * 16;
      const __bf16* bcol = kbh + (size_t)(jbase + l16) * DHH;
      v16bf b0 = make_b_bf16(bcol, hi, 0, 1);
      v16bf b1 = make_b_bf16(bcol, hi, 32, 1);
      v8f acc = {};
      acc = wmma_bf16(qa0, b0, acc);
      acc = wmma_bf16(qa1, b1, acc);
#pragma unroll
      for (int r = 0; r < 8; ++r)
        S[(rbase + r + 8 * hi) * NN + jbase + l16] += acc[r] * blend;
    }
  }
  __syncthreads();

  // ---- attention softmax ----------------------------------------------------
  softmax_rows_32x1024(S, 1.0f, 0.0f);

  // V panel must be resident in LDS before phase C
#if HAS_TDM
  if (wave == 0) __builtin_amdgcn_s_wait_tensorcnt(0);
#endif
  __syncthreads();

  // ---- Phase C: O = attn @ V (32x1024 . 1024x64) ---------------------------
  {
    const int dbase = q4 * 16;
    const float* srow = S + (rbase + l16) * NN;
#if HAS_TDM
    const __bf16* vcol = Vl + dbase + l16;       // LDS V panel
#else
    const __bf16* vcol = vbh + dbase + l16;      // global fallback
#endif
    v8f acc = {};
#pragma unroll 4
    for (int kb = 0; kb < 32; ++kb) {
      v16bf a = make_a_f32(srow, hi, kb * 32);   // LDS f32 -> bf16 fragment
      v16bf b = make_b_bf16(vcol, hi, kb * 32, DHH);
      acc = wmma_bf16(a, b, acc);
    }
#pragma unroll
    for (int r = 0; r < 8; ++r)
      O[(rbase + r + 8 * hi) * DHH + dbase + l16] = acc[r];
  }
  __syncthreads();

  // ---- Phase D: per-head merge, out = O @ w_merge^T + b_merge --------------
  {
    const int dbase = q4 * 16;
    const float* orow = O + (rbase + l16) * DHH;
    const __bf16* wrow = wmb + (size_t)(dbase + l16) * DHH;
    v16bf a0 = make_a_f32(orow, hi, 0);
    v16bf a1 = make_a_f32(orow, hi, 32);
    v16bf b0 = make_b_bf16(wrow, hi, 0, 1);
    v16bf b1 = make_b_bf16(wrow, hi, 32, 1);
    v8f m = {};
    m = wmma_bf16(a0, b0, m);
    m = wmma_bf16(a1, b1, m);
    const float bias = bmerge[dbase + l16];
    const int bidx = bh >> 3, h = bh & 7;
#pragma unroll
    for (int r = 0; r < 8; ++r) {
      int nrow = i0 + rbase + r + 8 * hi;
      int col  = h * DHH + dbase + l16;
      merged[(size_t)(bidx * NN + nrow) * INNR + col] = (__bf16)(m[r] + bias);
    }
  }
}

// ---------------------------------------------------------------------------
// Kernel 4: output projection.  out[m,c] = sum_k merged[m,k]*w_out[c,k] + b_out[c]
// ---------------------------------------------------------------------------
__global__ void out_proj_kernel(const __bf16* __restrict__ merged, const __bf16* __restrict__ wob,
                                const float* __restrict__ bout, float* __restrict__ out) {
  const int wid  = blockIdx.x * 8 + (threadIdx.x >> 5);
  const int lane = threadIdx.x & 31;
  const int hi   = lane >> 4;
  const int l16  = lane & 15;
  const int tm   = wid >> 5;        // 512 row tiles
  const int tn   = wid & 31;        // 32 col tiles

  const __bf16* arow = merged + (size_t)(tm * 16 + l16) * INNR;
  const __bf16* brow = wob    + (size_t)(tn * 16 + l16) * INNR;

  v8f acc = {};
#pragma unroll 4
  for (int kk = 0; kk < INNR; kk += 32) {
    if (kk + 64 < INNR) __builtin_prefetch(arow + kk + 64, 0, 1);
    v16bf a = make_a_bf16(arow, hi, kk);
    v16bf b = make_b_bf16(brow, hi, kk, 1);
    acc = wmma_bf16(a, b, acc);
  }

  const int c = tn * 16 + l16;
  const float bias = bout[c];
#pragma unroll
  for (int r = 0; r < 8; ++r)
    out[(size_t)(tm * 16 + r + 8 * hi) * DIMM + c] = acc[r] + bias;
}

// ---------------------------------------------------------------------------
// Host launcher
// ---------------------------------------------------------------------------
extern "C" void kernel_launch(void* const* d_in, const int* in_sizes, int n_in,
                              void* d_out, int out_size, void* d_ws, size_t ws_size,
                              hipStream_t stream) {
  (void)in_sizes; (void)n_in; (void)out_size; (void)ws_size;

  const float* x      = (const float*)d_in[0];
  const float* lidar  = (const float*)d_in[1];
  const float* w_qkv  = (const float*)d_in[2];
  const float* w_mrg  = (const float*)d_in[3];
  const float* b_mrg  = (const float*)d_in[4];
  const float* w_out  = (const float*)d_in[5];
  const float* b_out  = (const float*)d_in[6];
  const float* conv_w = (const float*)d_in[7];
  const float* conv_b = (const float*)d_in[8];

  char* p = (char*)d_ws;
  __bf16* wqb    = (__bf16*)p; p += (size_t)WQ_ELEMS * 2;
  __bf16* wmb    = (__bf16*)p; p += (size_t)WM_ELEMS * 2;
  __bf16* wob    = (__bf16*)p; p += (size_t)WO_ELEMS * 2;
  const size_t headElems = (size_t)BB * HH * NN * DHH;   // 4,194,304
  __bf16* qh     = (__bf16*)p; p += headElems * 2;
  __bf16* kh     = (__bf16*)p; p += headElems * 2;
  __bf16* vh     = (__bf16*)p; p += headElems * 2;
  __bf16* lidh   = (__bf16*)p; p += headElems * 2;
  __bf16* merged = (__bf16*)p; p += (size_t)MROWS * INNR * 2;

  cvt_weights_kernel<<<512, 256, 0, stream>>>(w_qkv, w_mrg, w_out, wqb, wmb, wob);
  split_lidar_kernel<<<4096, 256, 0, stream>>>(lidar, lidh);
  qkv_gemm_kernel<<<6144, 256, 0, stream>>>(x, wqb, qh, kh, vh);

  const int smemBytes = ATTN_SMEM_BYTES;   // 278528 (fixed layout, <320KB/WGP)
  (void)hipFuncSetAttribute(reinterpret_cast<const void*>(attn_kernel),
                            hipFuncAttributeMaxDynamicSharedMemorySize, smemBytes);
  attn_kernel<<<BB * HH * (NN / 32), 256, smemBytes, stream>>>(
      qh, kh, vh, lidh, wmb, b_mrg, conv_w, conv_b, merged);

  out_proj_kernel<<<2048, 256, 0, stream>>>(merged, wob, b_out, (float*)d_out);

  (void)hipMemcpyAsync((float*)d_out + (size_t)MROWS * DIMM, lidar,
                       (size_t)BB * NN * INNR * sizeof(float), hipMemcpyDeviceToDevice, stream);
}
